// GraphCAD_16621523436245
// MI455X (gfx1250) — matrix operand order repaired and checked by reading, not compiled, not run on hardware
//
#include <hip/hip_runtime.h>
#include <hip/hip_bf16.h>
#include <math.h>

typedef __attribute__((ext_vector_type(16))) _Float16 v16h;
typedef __attribute__((ext_vector_type(8)))  float    v8f;
typedef __attribute__((ext_vector_type(4)))  int      v4i;

union AF { v16h v; unsigned u[8]; _Float16 h[16]; };

#define NNODE 8000
#define NF    128
#define KP0   416   // 400 padded to 13*32
#define KP1   32    // 20 padded to 32

#define ACT_NONE  0
#define ACT_RELU  1
#define ACT_PRELU 2

// ---- gfx1250 async global->LDS copy (guarded; fallback = plain copies) ----
#if defined(__gfx1250__) && __has_builtin(__builtin_amdgcn_global_load_async_to_lds_b128)
#define HAVE_ASYNC_LDS 1
typedef __attribute__((address_space(1))) v4i as1_v4i;
typedef __attribute__((address_space(3))) v4i as3_v4i;
__device__ __forceinline__ void async_g2l_b128(const void* g, void* l) {
  __builtin_amdgcn_global_load_async_to_lds_b128((as1_v4i*)g, (as3_v4i*)l, 0, 0);
}
__device__ __forceinline__ void wait_async_lds() {
#if __has_builtin(__builtin_amdgcn_s_wait_asynccnt)
  __builtin_amdgcn_s_wait_asynccnt(0);
#else
  asm volatile("s_wait_asynccnt 0x0" ::: "memory");
#endif
}
#endif

// ---------------- workspace layout (float element offsets) ----------------
enum : size_t {
  OFF_PROP0 = 0,                          // 8000*128
  OFF_PROP1 = OFF_PROP0 + 1024000,        // 8000*128
  OFF_HBUF  = OFF_PROP1 + 1024000,        // 8000*128
  OFF_S0    = OFF_HBUF  + 1024000,        // 8000*400 (logits0 -> softmax in place)
  OFF_CORR0 = OFF_S0    + 3200000,        // 8000*128
  OFF_XC1   = OFF_CORR0 + 1024000,        // 400*128
  OFF_T0    = OFF_XC1   + 51200,          // 400*128
  OFF_CORR1 = OFF_T0    + 51200,          // 400*128
  OFF_HC1   = OFF_CORR1 + 51200,          // 400*128
  OFF_S1    = OFF_HC1   + 51200,          // 400*20
  OFF_XC2   = OFF_S1    + 8000,           // 20*128
  OFF_T1    = OFF_XC2   + 2560,           // 20*128
  OFF_CORR2 = OFF_T1    + 2560,           // 20*128
  OFF_P1    = OFF_CORR2 + 2560,           // 8000*20
  OFF_CS0   = OFF_P1    + 160000,         // 400
  OFF_CS1   = OFF_CS0   + 400,            // 32
  OFF_STAT  = OFF_CS1   + 32,             // 256 (mean[128] + cnorm[128], reused)
  OFF_GG    = OFF_STAT  + 256,            // f16: 128*416 + 128*32 halves (16B aligned)
};  // total ~7.70M floats ~= 31 MB

// =======================================================================
// Generic tiled WMMA GEMM: C[M,N] = act(A @ B + bias)
// A: [M,K] (transA=0)  or logical A[m,k]=A[k*M+m] (transA=1, A stored [K,M])
// B: [K,N] row-major.  Block = 128 thr = 4 waves; tile 32x64; each wave 16x32.
// Fast unguarded tile-load paths when the tile is fully in bounds.
// =======================================================================
__global__ __launch_bounds__(128)
void gemm_wmma_kernel(const float* __restrict__ A, const float* __restrict__ B,
                      const float* __restrict__ bias, const float* __restrict__ prelu,
                      float* __restrict__ C, int M, int N, int K, int transA, int act)
{
  __shared__ _Float16 As[32][32];   // m-major, k contiguous
  __shared__ _Float16 Bs[64][32];   // n-major, k contiguous
  const int tid  = threadIdx.x;
  const int lane = tid & 31;
  const int wave = tid >> 5;
  const int wm = (wave & 1) * 16;
  const int wn = (wave >> 1) * 32;
  const int m0 = blockIdx.x * 32;
  const int n0 = blockIdx.y * 64;
  const int nlo = lane & 15;
  const int b0  = (lane < 16) ? 0 : 8;    // A-frag K base / C-frag M offset
  const int kbB = (lane < 16) ? 0 : 16;   // unused (B-frag K base is b0)
  const bool fullM = (m0 + 32 <= M);
  const bool fullN = (n0 + 64 <= N);

  v8f acc0 = {}; v8f acc1 = {};

  const int ksteps = (K + 31) >> 5;
  for (int ks = 0; ks < ksteps; ++ks) {
    const int k0 = ks << 5;
    const bool fullK = (k0 + 32 <= K);

    // ---- A tile ----
    if (!transA) {
      if (fullM & fullK) {
        #pragma unroll
        for (int i = 0; i < 8; ++i) {
          int e = tid + i * 128, m = e >> 5, k = e & 31;
          As[m][k] = (_Float16)A[(size_t)(m0 + m) * K + (k0 + k)];
        }
      } else {
        #pragma unroll
        for (int i = 0; i < 8; ++i) {
          int e = tid + i * 128, m = e >> 5, k = e & 31;
          float v = 0.f;
          if ((m0 + m) < M && (k0 + k) < K) v = A[(size_t)(m0 + m) * K + (k0 + k)];
          As[m][k] = (_Float16)v;
        }
      }
    } else {
      if (fullM & fullK) {
        #pragma unroll
        for (int i = 0; i < 8; ++i) {
          int e = tid + i * 128, m = e & 31, k = e >> 5;
          As[m][k] = (_Float16)A[(size_t)(k0 + k) * M + (m0 + m)];
        }
      } else {
        #pragma unroll
        for (int i = 0; i < 8; ++i) {
          int e = tid + i * 128, m = e & 31, k = e >> 5;
          float v = 0.f;
          if ((m0 + m) < M && (k0 + k) < K) v = A[(size_t)(k0 + k) * M + (m0 + m)];
          As[m][k] = (_Float16)v;
        }
      }
    }
    // ---- B tile (n-fast coalesced reads) ----
    if (fullN & fullK) {
      #pragma unroll
      for (int i = 0; i < 16; ++i) {
        int e = tid + i * 128, n = e & 63, k = e >> 6;
        Bs[n][k] = (_Float16)B[(size_t)(k0 + k) * N + (n0 + n)];
      }
    } else {
      #pragma unroll
      for (int i = 0; i < 16; ++i) {
        int e = tid + i * 128, n = e & 63, k = e >> 6;
        float v = 0.f;
        if ((n0 + n) < N && (k0 + k) < K) v = B[(size_t)(k0 + k) * N + (n0 + n)];
        Bs[n][k] = (_Float16)v;
      }
    }
    __syncthreads();

    AF a;
    {
      int m = wm + nlo;
      #pragma unroll
      for (int j = 0; j < 8; ++j) {
        int k = b0 + ((j < 4) ? (2 * j) : (8 + 2 * j));
        a.u[j] = *(const unsigned*)&As[m][k];
      }
    }
    #pragma unroll
    for (int t = 0; t < 2; ++t) {
      AF b;
      int n = wn + t * 16 + nlo;
      #pragma unroll
      for (int j = 0; j < 8; ++j)
        b.u[j] = *(const unsigned*)&Bs[n][b0 + ((j < 4) ? (2 * j) : (8 + 2 * j))];
      if (t == 0)
        acc0 = __builtin_amdgcn_wmma_f32_16x16x32_f16(false, a.v, false, b.v, (short)0, acc0, false, false);
      else
        acc1 = __builtin_amdgcn_wmma_f32_16x16x32_f16(false, a.v, false, b.v, (short)0, acc1, false, false);
    }
    __syncthreads();
  }

  const float slope = (act == ACT_PRELU && prelu) ? prelu[0] : 0.f;
  #pragma unroll
  for (int t = 0; t < 2; ++t) {
    v8f accv = t ? acc1 : acc0;
    int col = n0 + wn + t * 16 + nlo;
    #pragma unroll
    for (int v = 0; v < 8; ++v) {
      int row = m0 + wm + v + b0;
      if (row < M && col < N) {
        float val = accv[v];
        if (bias) val += bias[col];
        if (act == ACT_RELU)       val = val > 0.f ? val : 0.f;
        else if (act == ACT_PRELU) val = val >= 0.f ? val : slope * val;
        C[(size_t)row * N + col] = val;
      }
    }
  }
}

// ---------------- BatchNorm (per-feature, training stats) ----------------
__global__ __launch_bounds__(256)
void bn_kernel(const float* __restrict__ x, const float* __restrict__ gamma,
               const float* __restrict__ beta, float* __restrict__ out, int M)
{
  __shared__ float s1[256], s2[256];
  const int f = blockIdx.x, tid = threadIdx.x;
  float sum = 0.f, sq = 0.f;
  for (int n = tid; n < M; n += 256) { float v = x[(size_t)n * NF + f]; sum += v; sq += v * v; }
  s1[tid] = sum; s2[tid] = sq; __syncthreads();
  for (int s = 128; s > 0; s >>= 1) {
    if (tid < s) { s1[tid] += s1[tid + s]; s2[tid] += s2[tid + s]; }
    __syncthreads();
  }
  const float mean = s1[0] / M;
  const float var  = s2[0] / M - mean * mean;
  const float rstd = rsqrtf(var + 1e-5f);
  const float g = gamma[f], b = beta[f];
  for (int n = tid; n < M; n += 256)
    out[(size_t)n * NF + f] = (x[(size_t)n * NF + f] - mean) * rstd * g + b;
}

// ---------------- column mean + centered L2 norm (for _node_corr) --------
__global__ __launch_bounds__(256)
void colstats_kernel(const float* __restrict__ Z, float* __restrict__ mean,
                     float* __restrict__ cnorm, int M)
{
  __shared__ float s1[256], s2[256];
  const int f = blockIdx.x, tid = threadIdx.x;
  float sum = 0.f, sq = 0.f;
  for (int n = tid; n < M; n += 256) { float v = Z[(size_t)n * NF + f]; sum += v; sq += v * v; }
  s1[tid] = sum; s2[tid] = sq; __syncthreads();
  for (int s = 128; s > 0; s >>= 1) {
    if (tid < s) { s1[tid] += s1[tid + s]; s2[tid] += s2[tid + s]; }
    __syncthreads();
  }
  if (tid == 0) {
    float mu = s1[0] / M;
    mean[f]  = mu;
    cnorm[f] = sqrtf(fmaxf(s2[0] - (float)M * mu * mu, 0.f));
  }
}

// ---------------- C[i,f] = zc[i,f] * rowsum(zc[i,:]) ---------------------
__global__ __launch_bounds__(128)
void node_corr_kernel(const float* __restrict__ Z, const float* __restrict__ mean,
                      const float* __restrict__ cnorm, float* __restrict__ out)
{
  __shared__ float sred[128];
  const int row = blockIdx.x, f = threadIdx.x;
  const float zc = (Z[(size_t)row * NF + f] - mean[f]) / (cnorm[f] + 1e-12f);
  sred[f] = zc; __syncthreads();
  for (int s = 64; s > 0; s >>= 1) { if (f < s) sred[f] += sred[f + s]; __syncthreads(); }
  out[(size_t)row * NF + f] = zc * sred[0];
}

// ---------------- row softmax (in place) ---------------------------------
__global__ __launch_bounds__(128)
void softmax_rows_kernel(float* __restrict__ X, int N)
{
  __shared__ float sred[128];
  const int tid = threadIdx.x;
  float* xr = X + (size_t)blockIdx.x * N;
  float mx = -1e30f;
  for (int j = tid; j < N; j += 128) mx = fmaxf(mx, xr[j]);
  sred[tid] = mx; __syncthreads();
  for (int s = 64; s > 0; s >>= 1) { if (tid < s) sred[tid] = fmaxf(sred[tid], sred[tid + s]); __syncthreads(); }
  mx = sred[0]; __syncthreads();
  float sum = 0.f;
  for (int j = tid; j < N; j += 128) sum += expf(xr[j] - mx);
  sred[tid] = sum; __syncthreads();
  for (int s = 64; s > 0; s >>= 1) { if (tid < s) sred[tid] += sred[tid + s]; __syncthreads(); }
  const float inv = 1.f / sred[0];
  __syncthreads();
  for (int j = tid; j < N; j += 128) xr[j] = expf(xr[j] - mx) * inv;
}

// ---------------- column sums --------------------------------------------
__global__ __launch_bounds__(128)
void colsum_kernel(const float* __restrict__ X, float* __restrict__ out, int M, int N)
{
  __shared__ float sred[128];
  const int col = blockIdx.x, tid = threadIdx.x;
  float sum = 0.f;
  for (int m = tid; m < M; m += 128) sum += X[(size_t)m * N + col];
  sred[tid] = sum; __syncthreads();
  for (int s = 64; s > 0; s >>= 1) { if (tid < s) sred[tid] += sred[tid + s]; __syncthreads(); }
  if (tid == 0) out[col] = sred[0];
}

// ---------------- X[m,n] /= (den[m] + 1e-12) -----------------------------
__global__ void scale_rows_kernel(float* __restrict__ X, const float* __restrict__ den,
                                  int M, int N)
{
  int i = blockIdx.x * 256 + threadIdx.x;
  if (i < M * N) X[i] = X[i] / (den[i / N] + 1e-12f);
}

// ------- ggT[f][k] = (sigmoid(corrNext[k,f]-T[k,f]))^2, f16, K padded ----
__global__ void ggprep_kernel(const float* __restrict__ corrNext, const float* __restrict__ T,
                              _Float16* __restrict__ ggT, int Mc, int Kpad)
{
  int idx = blockIdx.x * 256 + threadIdx.x;
  if (idx >= NF * Kpad) return;
  int f = idx / Kpad, k = idx % Kpad;
  float val = 0.f;
  if (k < Mc) {
    float d = corrNext[(size_t)k * NF + f] - T[(size_t)k * NF + f];
    float g = 1.f / (1.f + expf(-d));
    val = g * g;
  }
  ggT[idx] = (_Float16)val;
}

// =======================================================================
// Fused edge kernel: per 16-edge tile, WMMA-GEMM the two edge masks
//   ms0 = (P1[r]*P1[c]) @ gg1   (K=20->32)
//   ms1 = (P0[r]*P0[c]) @ gg0   (K=400->416, B staged async into LDS)
// then per edge run the sequential feature scan producing adjs[f*E+e].
// Block = 64 threads (2 waves, 32 edges).  Both waves share the B tile.
// =======================================================================
__global__ __launch_bounds__(64)
void edge_adjs_kernel(const int* __restrict__ rows, const int* __restrict__ cols,
                      const float* __restrict__ adjv,
                      const float* __restrict__ P0, const float* __restrict__ P1,
                      const _Float16* __restrict__ ggT0, const _Float16* __restrict__ ggT1,
                      float* __restrict__ adjs, int E)
{
  __shared__ float ms0[2][16][NF];
  __shared__ float ms1[2][16][NF];
  __shared__ __attribute__((aligned(16))) _Float16 BsT[128][32];  // gg0 K-slice, n-major

  const int tid  = threadIdx.x;
  const int wave = tid >> 5, lane = tid & 31;
  const int nlo = lane & 15;
  const int b0  = (lane < 16) ? 0 : 8;
  const int kbB = (lane < 16) ? 0 : 16;
  int e = (blockIdx.x * 2 + wave) * 16 + nlo;
  if (e >= E) e = E - 1;
  const int r = rows[e], c = cols[e];

  const float* p0r = P0 + (size_t)r * 400;
  const float* p0c = P0 + (size_t)c * 400;
  __builtin_prefetch(p0r, 0, 0);      // global_prefetch_b8
  __builtin_prefetch(p0c, 0, 0);

  // ---- ms0: single K-step over padded 32 (valid 20), B read direct ----
  AF a0;
  {
    const float* pr = P1 + (size_t)r * 20;
    const float* pc = P1 + (size_t)c * 20;
    #pragma unroll
    for (int j = 0; j < 8; ++j) {
      int k = b0 + ((j < 4) ? 2 * j : 8 + 2 * j);
      float x0 = (k     < 20) ? pr[k]     * pc[k]     : 0.f;
      float x1 = (k + 1 < 20) ? pr[k + 1] * pc[k + 1] : 0.f;
      a0.h[2 * j]     = (_Float16)x0;
      a0.h[2 * j + 1] = (_Float16)x1;
    }
  }
  v8f zeroV = {};
  v8f acc0[8];
  #pragma unroll
  for (int t = 0; t < 8; ++t) {
    AF b;
    const _Float16* br = ggT1 + (size_t)(t * 16 + nlo) * KP1 + b0;
    #pragma unroll
    for (int j = 0; j < 8; ++j) b.u[j] = *(const unsigned*)&br[(j < 4) ? 2 * j : 8 + 2 * j];
    acc0[t] = __builtin_amdgcn_wmma_f32_16x16x32_f16(false, a0.v, false, b.v, (short)0, zeroV, false, false);
  }

  // ---- ms1: 13 K-steps over padded 416 (valid 400) ----
  v8f acc1[8];
  #pragma unroll
  for (int t = 0; t < 8; ++t) acc1[t] = zeroV;

  for (int ks = 0; ks < 13; ++ks) {
    const int k0 = ks * 32;

    // stage B K-slice into LDS: 128 rows x 32 halves (64 B each)
    #pragma unroll
    for (int i = 0; i < 2; ++i) {
      const int f = tid * 2 + i;
      const _Float16* srcp = ggT0 + (size_t)f * KP0 + k0;
#ifdef HAVE_ASYNC_LDS
      #pragma unroll
      for (int q = 0; q < 4; ++q)
        async_g2l_b128(srcp + q * 8, &BsT[f][q * 8]);
#else
      const float4* s4 = (const float4*)srcp;
      float4* d4 = (float4*)&BsT[f][0];
      d4[0] = s4[0]; d4[1] = s4[1]; d4[2] = s4[2]; d4[3] = s4[3];
#endif
    }

    // build A fragment while the async copies are in flight
    AF a;
    if (k0 + 32 <= 400) {          // fully in-range: guard-free pair loads
      #pragma unroll
      for (int j = 0; j < 8; ++j) {
        int k = k0 + b0 + ((j < 4) ? 2 * j : 8 + 2 * j);
        float2 r2 = *(const float2*)&p0r[k];
        float2 c2 = *(const float2*)&p0c[k];
        a.h[2 * j]     = (_Float16)(r2.x * c2.x);
        a.h[2 * j + 1] = (_Float16)(r2.y * c2.y);
      }
    } else {                        // tail step (k0 = 384)
      #pragma unroll
      for (int j = 0; j < 8; ++j) {
        int k = k0 + b0 + ((j < 4) ? 2 * j : 8 + 2 * j);
        float x0 = 0.f, x1 = 0.f;
        if (k + 1 < 400) { x0 = p0r[k] * p0c[k]; x1 = p0r[k + 1] * p0c[k + 1]; }
        else if (k < 400) { x0 = p0r[k] * p0c[k]; }
        a.h[2 * j]     = (_Float16)x0;
        a.h[2 * j + 1] = (_Float16)x1;
      }
    }

#ifdef HAVE_ASYNC_LDS
    wait_async_lds();
#endif
    __syncthreads();

    #pragma unroll
    for (int t = 0; t < 8; ++t) {
      AF b;
      const _Float16* br = &BsT[t * 16 + nlo][b0];
      #pragma unroll
      for (int j = 0; j < 8; ++j) b.u[j] = *(const unsigned*)&br[(j < 4) ? 2 * j : 8 + 2 * j];
      acc1[t] = __builtin_amdgcn_wmma_f32_16x16x32_f16(false, a.v, false, b.v, (short)0, acc1[t], false, false);
    }
    __syncthreads();   // before next slice overwrites BsT
  }

  // ---- dump C tiles to LDS (M=edge, N=feature) ----
  #pragma unroll
  for (int t = 0; t < 8; ++t) {
    int f = t * 16 + nlo;
    #pragma unroll
    for (int v = 0; v < 8; ++v) {
      int ml = v + b0;
      ms0[wave][ml][f] = acc0[t][v];
      ms1[wave][ml][f] = acc1[t][v];
    }
  }
  __syncthreads();

  // ---- per-edge sequential feature scan (carry product) ----
  if (tid < 32) {
    int e2 = blockIdx.x * 32 + tid;
    if (e2 < E) {
      const int w2 = tid >> 4, m2 = tid & 15;
      float a = adjv[e2];
      for (int f = 0; f < NF; ++f) {
        float t0 = a * ms0[w2][m2][f];
        a = t0 * ms1[w2][m2][f];
        adjs[(size_t)f * E + e2] = t0 + a;   // temp0 + temp1
      }
    }
  }
}

// ---------------- diffusion: out[r] += nv[e] * x[c] ----------------------
__global__ void zero_kernel(float* __restrict__ p, int n)
{
  int i = blockIdx.x * 256 + threadIdx.x;
  if (i < n) p[i] = 0.f;
}

__global__ void spmm_kernel(const int* __restrict__ rows, const int* __restrict__ cols,
                            const float* __restrict__ nv, const float* __restrict__ xin,
                            float* __restrict__ xout, int E)
{
  int t = blockIdx.x * 256 + threadIdx.x;
  int e = t >> 5;
  if (e >= E) return;
  int q = t & 31;
  const int ci = cols[e], ri = rows[e];
  const float w = nv[e];
  const float4 v = ((const float4*)(xin + (size_t)ci * NF))[q];
  float* o = xout + (size_t)ri * NF + q * 4;
  atomicAdd(o + 0, w * v.x);
  atomicAdd(o + 1, w * v.y);
  atomicAdd(o + 2, w * v.z);
  atomicAdd(o + 3, w * v.w);
}

// ---------------- final layer + log_softmax (N classes = 2) --------------
__global__ __launch_bounds__(128)
void mlp_out_kernel(const float* __restrict__ H, const float* __restrict__ W3,
                    const float* __restrict__ b3, float* __restrict__ out)
{
  __shared__ float s0[128], s1[128];
  const int n = blockIdx.x, f = threadIdx.x;
  const float v = H[(size_t)n * NF + f];
  s0[f] = v * W3[f * 2 + 0];
  s1[f] = v * W3[f * 2 + 1];
  __syncthreads();
  for (int s = 64; s > 0; s >>= 1) {
    if (f < s) { s0[f] += s0[f + s]; s1[f] += s1[f + s]; }
    __syncthreads();
  }
  if (f == 0) {
    float l0 = s0[0] + b3[0], l1 = s1[0] + b3[1];
    float mx = fmaxf(l0, l1);
    float lse = mx + logf(expf(l0 - mx) + expf(l1 - mx));
    out[n * 2 + 0] = l0 - lse;
    out[n * 2 + 1] = l1 - lse;
  }
}

// =======================================================================
extern "C" void kernel_launch(void* const* d_in, const int* in_sizes, int n_in,
                              void* d_out, int out_size, void* d_ws, size_t ws_size,
                              hipStream_t stream)
{
  (void)n_in; (void)out_size; (void)ws_size;
  const float* x     = (const float*)d_in[0];
  const float* x_cov = (const float*)d_in[1];
  const int*   erow  = (const int*)d_in[2];
  const int*   ecol  = (const int*)d_in[3];
  const float* adjv  = (const float*)d_in[4];
  const float* nvls  = (const float*)d_in[5];
  const float* gamma = (const float*)d_in[6];
  const float* beta  = (const float*)d_in[7];
  const float* c0W1  = (const float*)d_in[8];
  const float* c0b1  = (const float*)d_in[9];
  const float* c0W2  = (const float*)d_in[10];
  const float* c0b2  = (const float*)d_in[11];
  const float* c1W1  = (const float*)d_in[12];
  const float* c1b1  = (const float*)d_in[13];
  const float* c1W2  = (const float*)d_in[14];
  const float* c1b2  = (const float*)d_in[15];
  // d_in[16..19] = level-2 classifier: dead code w.r.t. outputs
  const float* mW1 = (const float*)d_in[20];
  const float* mb1 = (const float*)d_in[21];
  const float* a1  = (const float*)d_in[22];
  const float* mW2 = (const float*)d_in[23];
  const float* mb2 = (const float*)d_in[24];
  const float* a2  = (const float*)d_in[25];
  const float* mW3 = (const float*)d_in[26];
  const float* mb3 = (const float*)d_in[27];
  const int E = in_sizes[2];

  float* ws  = (float*)d_ws;
  float* out = (float*)d_out;
  float* adjs_out = out + (size_t)NNODE * 2;

  float* PROP0 = ws + OFF_PROP0;
  float* PROP1 = ws + OFF_PROP1;
  float* HBUF  = ws + OFF_HBUF;
  float* S0    = ws + OFF_S0;
  float* CORR0 = ws + OFF_CORR0;
  float* XC1   = ws + OFF_XC1;
  float* T0    = ws + OFF_T0;
  float* CORR1 = ws + OFF_CORR1;
  float* HC1   = ws + OFF_HC1;
  float* S1    = ws + OFF_S1;
  float* XC2   = ws + OFF_XC2;
  float* T1    = ws + OFF_T1;
  float* CORR2 = ws + OFF_CORR2;
  float* P1    = ws + OFF_P1;
  float* CS0   = ws + OFF_CS0;
  float* CS1   = ws + OFF_CS1;
  float* MEAN  = ws + OFF_STAT;
  float* CNORM = ws + OFF_STAT + 128;
  _Float16* GGT0 = (_Float16*)(ws + OFF_GG);
  _Float16* GGT1 = GGT0 + (size_t)NF * KP0;

  #define GEMM(A_, B_, BIAS_, PRELU_, C_, M_, N_, K_, TA_, ACT_)                         \
    gemm_wmma_kernel<<<dim3(((M_) + 31) / 32, ((N_) + 63) / 64), 128, 0, stream>>>(      \
        A_, B_, BIAS_, PRELU_, C_, M_, N_, K_, TA_, ACT_)

  // 1. BatchNorm(x) -> propagation seed
  bn_kernel<<<NF, 256, 0, stream>>>(x, gamma, beta, PROP0, NNODE);

  // 2. corr0 = node_corr(x_cov)
  colstats_kernel<<<NF, 256, 0, stream>>>(x_cov, MEAN, CNORM, NNODE);
  node_corr_kernel<<<NNODE, 128, 0, stream>>>(x_cov, MEAN, CNORM, CORR0);

  // 3. Level-0 classifier -> S0 [8000,400]
  GEMM(x_cov, c0W1, c0b1, nullptr, HBUF, NNODE, 128, 128, 0, ACT_RELU);
  GEMM(HBUF, c0W2, c0b2, nullptr, S0, NNODE, 400, 128, 0, ACT_NONE);
  softmax_rows_kernel<<<NNODE, 128, 0, stream>>>(S0, 400);
  colsum_kernel<<<400, 128, 0, stream>>>(S0, CS0, NNODE, 400);

  // 4. Pool: XC1 = (S0^T @ x_cov) / colsum ; T0 = S0^T @ corr0
  GEMM(S0, x_cov, nullptr, nullptr, XC1, 400, 128, NNODE, 1, ACT_NONE);
  scale_rows_kernel<<<(400 * 128 + 255) / 256, 256, 0, stream>>>(XC1, CS0, 400, 128);
  GEMM(S0, CORR0, nullptr, nullptr, T0, 400, 128, NNODE, 1, ACT_NONE);

  // 5. corr1 = node_corr(XC1); gg0 = sigmoid(corr1 - T0)^2 (f16, K padded)
  colstats_kernel<<<NF, 256, 0, stream>>>(XC1, MEAN, CNORM, 400);
  node_corr_kernel<<<400, 128, 0, stream>>>(XC1, MEAN, CNORM, CORR1);
  ggprep_kernel<<<(NF * KP0 + 255) / 256, 256, 0, stream>>>(CORR1, T0, GGT0, 400, KP0);

  // 6. Level-1 classifier -> S1 [400,20]
  GEMM(XC1, c1W1, c1b1, nullptr, HC1, 400, 128, 128, 0, ACT_RELU);
  GEMM(HC1, c1W2, c1b2, nullptr, S1, 400, 20, 128, 0, ACT_NONE);
  softmax_rows_kernel<<<400, 128, 0, stream>>>(S1, 20);
  colsum_kernel<<<20, 128, 0, stream>>>(S1, CS1, 400, 20);

  // 7. Pool level 1: XC2, T1; corr2; gg1
  GEMM(S1, XC1, nullptr, nullptr, XC2, 20, 128, 400, 1, ACT_NONE);
  scale_rows_kernel<<<(20 * 128 + 255) / 256, 256, 0, stream>>>(XC2, CS1, 20, 128);
  GEMM(S1, CORR1, nullptr, nullptr, T1, 20, 128, 400, 1, ACT_NONE);
  colstats_kernel<<<NF, 256, 0, stream>>>(XC2, MEAN, CNORM, 20);
  node_corr_kernel<<<20, 128, 0, stream>>>(XC2, MEAN, CNORM, CORR2);
  ggprep_kernel<<<(NF * KP1 + 255) / 256, 256, 0, stream>>>(CORR2, T1, GGT1, 20, KP1);

  // 8. P1 = S0 @ S1  [8000,20]
  GEMM(S0, S1, nullptr, nullptr, P1, NNODE, 20, 400, 0, ACT_NONE);

  // 9. Fused edge-mask GEMMs + per-feature scan -> adjs [128, E]
  edge_adjs_kernel<<<(E + 31) / 32, 64, 0, stream>>>(erow, ecol, adjv, S0, P1,
                                                     GGT0, GGT1, adjs_out, E);

  // 10. Diffusion: 10 ping-pong SpMM steps (alpha = 0)
  float* src = PROP0;
  float* dst = PROP1;
  const int nxf = NNODE * NF;
  const int spmm_blocks = (int)(((size_t)E * 32 + 255) / 256);
  for (int it = 0; it < 10; ++it) {
    zero_kernel<<<(nxf + 255) / 256, 256, 0, stream>>>(dst, nxf);
    spmm_kernel<<<spmm_blocks, 256, 0, stream>>>(erow, ecol, nvls, src, dst, E);
    float* tmp = src; src = dst; dst = tmp;
  }
  // after 10 iters result is in src (== PROP0)

  // 11. MLP head + log_softmax
  GEMM(src, mW1, mb1, a1, HBUF, NNODE, 128, 128, 0, ACT_PRELU);
  GEMM(HBUF, mW2, mb2, a2, dst, NNODE, 128, 128, 0, ACT_PRELU);
  mlp_out_kernel<<<NNODE, 128, 0, stream>>>(dst, mW3, mb3, out);

  #undef GEMM
}
